// MultiHeadSelfAttention_60301340836395
// MI455X (gfx1250) — matrix operand order repaired and checked by reading, not compiled, not run on hardware
//
#include <hip/hip_runtime.h>
#include <hip/hip_bf16.h>
#include <stdint.h>

#define BATCH 4
#define SEQ   2048
#define EMB   768
#define NH    12
#define HD    64
#define MTOT  (BATCH * SEQ)   // 8192
#define NQKV  (3 * EMB)       // 2304
#define ATTN_SCALE 0.125f     // 64^-0.5

typedef __bf16 v16bf __attribute__((ext_vector_type(16)));
typedef __bf16 bf16x8 __attribute__((ext_vector_type(8)));
typedef float  v8f    __attribute__((ext_vector_type(8)));

__device__ __forceinline__ __bf16 f2bf(float f) {
  unsigned u = __builtin_bit_cast(unsigned, f);
  u += 0x7FFFu + ((u >> 16) & 1u);           // round-to-nearest-even
  unsigned short h = (unsigned short)(u >> 16);
  return __builtin_bit_cast(__bf16, h);
}

__device__ __forceinline__ v8f zero8() {
  v8f z = {0.f, 0.f, 0.f, 0.f, 0.f, 0.f, 0.f, 0.f};
  return z;
}

// ---- WMMA fragment loads (ISA 7.12.2 layouts, wave32) -----------------------
// A (16x32 bf16, row-major source): lane half h, row r=lane&15:
//   elems 0-7  = K[k0 + h*8 .. +7],  elems 8-15 = K[k0 + 16 + h*8 .. +7]
__device__ __forceinline__ v16bf load_a_frag(const __bf16* __restrict__ A, int lda,
                                             int row0, int k0, int lane) {
  int half = lane >> 4, r = lane & 15;
  const __bf16* p = A + (size_t)(row0 + r) * lda + (k0 + half * 8);
  bf16x8 lo = *(const bf16x8*)p;
  bf16x8 hi = *(const bf16x8*)(p + 16);
  return __builtin_shufflevector(lo, hi, 0, 1, 2, 3, 4, 5, 6, 7,
                                 8, 9, 10, 11, 12, 13, 14, 15);
}

// B (32x16 bf16) loaded from N-major (transposed) memory Bt[N][K]:
//   lane n=lane&15 (column), kh=lane>>4: elems 0-7 = K[k0 + kh*8 .. +7], elems 8-15 = K[k0 + 16 + kh*8 .. +7]
__device__ __forceinline__ v16bf load_b_frag(const __bf16* __restrict__ Bt, int ldb,
                                             int n0, int k0, int lane) {
  int kh = lane >> 4, n = lane & 15;
  const __bf16* p = Bt + (size_t)(n0 + n) * ldb + (k0 + kh * 8);
  bf16x8 lo = *(const bf16x8*)p;
  bf16x8 hi = *(const bf16x8*)(p + 16);
  return __builtin_shufflevector(lo, hi, 0, 1, 2, 3, 4, 5, 6, 7,
                                 8, 9, 10, 11, 12, 13, 14, 15);
}

// Same B-fragment pattern but from an LDS tile (tile[N][ld] bf16, 16B-aligned rows)
__device__ __forceinline__ v16bf lds_b_frag(const __bf16* tile, int ld,
                                            int n0, int k0, int lane) {
  int kh = lane >> 4, n = lane & 15;
  const __bf16* p = tile + (n0 + n) * ld + (k0 + kh * 8);
  bf16x8 lo = *(const bf16x8*)p;
  bf16x8 hi = *(const bf16x8*)(p + 16);
  return __builtin_shufflevector(lo, hi, 0, 1, 2, 3, 4, 5, 6, 7,
                                 8, 9, 10, 11, 12, 13, 14, 15);
}

__device__ __forceinline__ v8f wmma_bf16(v16bf a, v16bf b, v8f c) {
  return __builtin_amdgcn_wmma_f32_16x16x32_bf16(false, a, false, b,
                                                 (short)0, c, false, false);
}

// ---- generic 64x64 wave-tile GEMM body --------------------------------------
template <int KDIM>
__device__ __forceinline__ void wave_gemm(const __bf16* __restrict__ A, int lda,
                                          const __bf16* __restrict__ Bt, int ldb,
                                          int m0, int n0, int lane, v8f acc[4][4]) {
#pragma unroll
  for (int mt = 0; mt < 4; ++mt)
#pragma unroll
    for (int nt = 0; nt < 4; ++nt) acc[mt][nt] = zero8();
#pragma unroll 1
  for (int kk = 0; kk < KDIM; kk += 32) {
    v16bf a[4], b[4];
#pragma unroll
    for (int t = 0; t < 4; ++t) a[t] = load_a_frag(A, lda, m0 + t * 16, kk, lane);
#pragma unroll
    for (int t = 0; t < 4; ++t) b[t] = load_b_frag(Bt, ldb, n0 + t * 16, kk, lane);
#pragma unroll
    for (int mt = 0; mt < 4; ++mt)
#pragma unroll
      for (int nt = 0; nt < 4; ++nt)
        acc[mt][nt] = wmma_bf16(a[mt], b[nt], acc[mt][nt]);
  }
}

// ---- prep kernels -----------------------------------------------------------
__global__ void convert_f32_bf16(const float* __restrict__ in,
                                 __bf16* __restrict__ out, int n) {
  int i = blockIdx.x * blockDim.x + threadIdx.x;
  if (i < n) out[i] = f2bf(in[i]);
}

// in: f32 [K rows][N cols] -> out: bf16 [N][K]
__global__ void transpose_f32_bf16(const float* __restrict__ in,
                                   __bf16* __restrict__ out, int K, int N) {
  int i = blockIdx.x * blockDim.x + threadIdx.x;
  if (i < K * N) {
    int n = i / K, k = i - n * K;
    out[(size_t)n * K + k] = f2bf(in[(size_t)k * N + n]);
  }
}

// ---- QKV projection: [8192,768] x [768,2304] + bias, scatter q/k/vT ---------
__global__ __launch_bounds__(256) void qkv_gemm_kernel(
    const __bf16* __restrict__ xbf, const __bf16* __restrict__ wqkvT,
    const float* __restrict__ bqkv, __bf16* __restrict__ qb,
    __bf16* __restrict__ kb, __bf16* __restrict__ vtb) {
  const int lane = threadIdx.x & 31, wave = threadIdx.x >> 5;
  const int m0 = blockIdx.y * 128 + (wave >> 2) * 64;
  const int n0 = blockIdx.x * 256 + (wave & 3) * 64;
  v8f acc[4][4];
  wave_gemm<EMB>(xbf, EMB, wqkvT, EMB, m0, n0, lane, acc);

  const int half = lane >> 4, lcol = lane & 15;
#pragma unroll
  for (int nt = 0; nt < 4; ++nt) {
    int n = n0 + nt * 16 + lcol;
    float bias = bqkv[n];
    int part = n / EMB;            // 0=q 1=k 2=v (16-wide tiles never straddle)
    int head = (n % EMB) / HD;
    int d = n % HD;
#pragma unroll
    for (int mt = 0; mt < 4; ++mt)
#pragma unroll
      for (int i = 0; i < 8; ++i) {
        int m = m0 + mt * 16 + half * 8 + i;
        int b = m >> 11, s = m & (SEQ - 1);
        int bh = b * NH + head;
        float val = acc[mt][nt][i] + bias;
        if (part == 0)
          qb[((size_t)bh * SEQ + s) * HD + d] = f2bf(val * ATTN_SCALE);
        else if (part == 1)
          kb[((size_t)bh * SEQ + s) * HD + d] = f2bf(val);
        else
          vtb[((size_t)bh * HD + d) * SEQ + s] = f2bf(val);  // V transposed
      }
  }
}

// ---- flash attention --------------------------------------------------------
// Block = 4 waves sharing one (b,h); each wave owns 32 q rows.
// K/V 64x64 tiles are staged cooperatively into LDS with
// GLOBAL_LOAD_ASYNC_TO_LDS_B128 (ASYNCcnt path), then consumed via ds_load.
__global__ __launch_bounds__(128) void attn_kernel(
    const __bf16* __restrict__ qb, const __bf16* __restrict__ kb,
    const __bf16* __restrict__ vtb, __bf16* __restrict__ aout) {
  __shared__ __bf16 plds[4][32][72] __attribute__((aligned(16)));  // per-wave P
  __shared__ __bf16 ktile[64][72] __attribute__((aligned(16)));    // [key][d]
  __shared__ __bf16 vtile[64][72] __attribute__((aligned(16)));    // [d][key]

  const int lane = threadIdx.x & 31, wave = threadIdx.x >> 5;
  const int half = lane >> 4, lcol = lane & 15;
  const int bh = blockIdx.x >> 4;                    // 16 q-tiles of 128 per (b,h)
  const int q0 = (blockIdx.x & 15) * 128 + wave * 32;
  const __bf16* Q  = qb  + (size_t)bh * SEQ * HD;
  const __bf16* Km = kb  + (size_t)bh * SEQ * HD;
  const __bf16* VT = vtb + (size_t)bh * HD * SEQ;

  v16bf qf[2][2];
#pragma unroll
  for (int mt = 0; mt < 2; ++mt)
#pragma unroll
    for (int kc = 0; kc < 2; ++kc)
      qf[mt][kc] = load_a_frag(Q, HD, q0 + mt * 16, kc * 32, lane);

  v8f o[2][4];
  float mrow[2][8], lrow[2][8];
#pragma unroll
  for (int mt = 0; mt < 2; ++mt) {
#pragma unroll
    for (int nt = 0; nt < 4; ++nt) o[mt][nt] = zero8();
#pragma unroll
    for (int i = 0; i < 8; ++i) { mrow[mt][i] = -3.0e38f; lrow[mt][i] = 0.f; }
  }

#pragma unroll 1
  for (int j0 = 0; j0 < SEQ; j0 += 64) {
    // --- cooperative async staging of K tile [64 keys][64 d] and
    //     V^T tile [64 d][64 keys] (512 x 16B chunks each, 128 threads) ------
#pragma unroll
    for (int it = 0; it < 4; ++it) {
      int chunk = threadIdx.x + it * 128;        // 0..511
      int row = chunk >> 3, colc = chunk & 7;    // 8 chunks of 8 bf16 per row
      unsigned lk = (unsigned)(uintptr_t)&ktile[row][colc * 8];
      unsigned gk = (unsigned)(((j0 + row) * HD + colc * 8) * sizeof(__bf16));
      asm volatile("global_load_async_to_lds_b128 %0, %1, %2"
                   :: "v"(lk), "v"(gk), "s"(Km) : "memory");
      unsigned lv = (unsigned)(uintptr_t)&vtile[row][colc * 8];
      unsigned gv = (unsigned)((row * SEQ + j0 + colc * 8) * sizeof(__bf16));
      asm volatile("global_load_async_to_lds_b128 %0, %1, %2"
                   :: "v"(lv), "v"(gv), "s"(VT) : "memory");
    }
    asm volatile("s_wait_asynccnt 0x0" ::: "memory");
    __syncthreads();

    // --- S = Q @ K^T  (32 x 64 logits), K B-frags from LDS ------------------
    v8f sfr[2][4];
#pragma unroll
    for (int mt = 0; mt < 2; ++mt)
#pragma unroll
      for (int nt = 0; nt < 4; ++nt) sfr[mt][nt] = zero8();
#pragma unroll
    for (int kc = 0; kc < 2; ++kc)
#pragma unroll
      for (int nt = 0; nt < 4; ++nt) {
        v16bf kf = lds_b_frag(&ktile[0][0], 72, nt * 16, kc * 32, lane);
#pragma unroll
        for (int mt = 0; mt < 2; ++mt)
          sfr[mt][nt] = wmma_bf16(qf[mt][kc], kf, sfr[mt][nt]);
      }
    // --- online softmax (row stats live in the lane owning the row) ---------
#pragma unroll
    for (int mt = 0; mt < 2; ++mt)
#pragma unroll
      for (int i = 0; i < 8; ++i) {
        float mx = sfr[mt][0][i];
#pragma unroll
        for (int nt = 1; nt < 4; ++nt) mx = fmaxf(mx, sfr[mt][nt][i]);
        mx = fmaxf(mx, __shfl_xor(mx, 1, 32));
        mx = fmaxf(mx, __shfl_xor(mx, 2, 32));
        mx = fmaxf(mx, __shfl_xor(mx, 4, 32));
        mx = fmaxf(mx, __shfl_xor(mx, 8, 32));
        float mnew = fmaxf(mrow[mt][i], mx);
        float alpha = __expf(mrow[mt][i] - mnew);
        float rs = 0.f;
#pragma unroll
        for (int nt = 0; nt < 4; ++nt) {
          float e = __expf(sfr[mt][nt][i] - mnew);
          sfr[mt][nt][i] = e;
          rs += e;
        }
        rs += __shfl_xor(rs, 1, 32);
        rs += __shfl_xor(rs, 2, 32);
        rs += __shfl_xor(rs, 4, 32);
        rs += __shfl_xor(rs, 8, 32);
        lrow[mt][i] = lrow[mt][i] * alpha + rs;
        mrow[mt][i] = mnew;
#pragma unroll
        for (int nt = 0; nt < 4; ++nt) o[mt][nt][i] *= alpha;
      }
    // --- stage P (C-layout) into per-wave LDS, reload as A-frags ------------
#pragma unroll
    for (int mt = 0; mt < 2; ++mt)
#pragma unroll
      for (int nt = 0; nt < 4; ++nt)
#pragma unroll
        for (int i = 0; i < 8; ++i)
          plds[wave][mt * 16 + half * 8 + i][nt * 16 + lcol] = f2bf(sfr[mt][nt][i]);
    __asm__ volatile("s_wait_dscnt 0" ::: "memory");
    // --- O += P @ V, V B-frags from LDS -------------------------------------
#pragma unroll
    for (int kc = 0; kc < 2; ++kc) {
      v16bf pa[2];
#pragma unroll
      for (int mt = 0; mt < 2; ++mt) {
        const __bf16* pl = &plds[wave][mt * 16 + lcol][kc * 32 + half * 8];
        bf16x8 lo = *(const bf16x8*)pl;
        bf16x8 hi = *(const bf16x8*)(pl + 16);
        pa[mt] = __builtin_shufflevector(lo, hi, 0, 1, 2, 3, 4, 5, 6, 7,
                                         8, 9, 10, 11, 12, 13, 14, 15);
      }
#pragma unroll
      for (int nt = 0; nt < 4; ++nt) {
        v16bf vf = lds_b_frag(&vtile[0][0], 72, nt * 16, kc * 32, lane);
#pragma unroll
        for (int mt = 0; mt < 2; ++mt)
          o[mt][nt] = wmma_bf16(pa[mt], vf, o[mt][nt]);
      }
    }
    __syncthreads();   // all waves done reading ktile/vtile before restaging
  }
  // --- normalize and write [B, S, H*D] bf16 ---------------------------------
  const int b = bh / NH, h = bh % NH;
#pragma unroll
  for (int mt = 0; mt < 2; ++mt)
#pragma unroll
    for (int nt = 0; nt < 4; ++nt)
#pragma unroll
      for (int i = 0; i < 8; ++i) {
        int m = q0 + mt * 16 + half * 8 + i;
        int d = nt * 16 + lcol;
        aout[((size_t)(b * SEQ + m)) * EMB + h * HD + d] =
            f2bf(o[mt][nt][i] / lrow[mt][i]);
      }
}

// ---- output projection: [8192,768] x [768,768] + bias -> f32 ----------------
__global__ __launch_bounds__(256) void proj_gemm_kernel(
    const __bf16* __restrict__ aout, const __bf16* __restrict__ wprojT,
    const float* __restrict__ bproj, float* __restrict__ out) {
  const int lane = threadIdx.x & 31, wave = threadIdx.x >> 5;
  const int m0 = blockIdx.y * 128 + (wave >> 2) * 64;
  const int n0 = blockIdx.x * 256 + (wave & 3) * 64;
  v8f acc[4][4];
  wave_gemm<EMB>(aout, EMB, wprojT, EMB, m0, n0, lane, acc);
  const int half = lane >> 4, lcol = lane & 15;
#pragma unroll
  for (int nt = 0; nt < 4; ++nt) {
    int n = n0 + nt * 16 + lcol;
    float bias = bproj[n];
#pragma unroll
    for (int mt = 0; mt < 4; ++mt)
#pragma unroll
      for (int i = 0; i < 8; ++i) {
        int m = m0 + mt * 16 + half * 8 + i;
        out[(size_t)m * EMB + n] = acc[mt][nt][i] + bias;
      }
  }
}

// ---- workspace layout (bytes, all 256-aligned) ------------------------------
#define WS_XBF    ((size_t)0)
#define WS_WQKVT  ((size_t)12582912)   // + 8192*768*2
#define WS_WPROJT ((size_t)16121856)   // + 2304*768*2
#define WS_QBF    ((size_t)17301504)   // + 768*768*2
#define WS_KBF    ((size_t)29884416)
#define WS_VTB    ((size_t)42467328)
#define WS_AOUT   ((size_t)55050240)   // end: 67633152

extern "C" void kernel_launch(void* const* d_in, const int* in_sizes, int n_in,
                              void* d_out, int out_size, void* d_ws, size_t ws_size,
                              hipStream_t stream) {
  (void)in_sizes; (void)n_in; (void)out_size; (void)ws_size;
  const float* x     = (const float*)d_in[0];
  const float* Wqkv  = (const float*)d_in[1];
  const float* bqkv  = (const float*)d_in[2];
  const float* Wproj = (const float*)d_in[3];
  const float* bproj = (const float*)d_in[4];
  float* out = (float*)d_out;
  char* ws = (char*)d_ws;

  __bf16* xbf    = (__bf16*)(ws + WS_XBF);
  __bf16* wqkvT  = (__bf16*)(ws + WS_WQKVT);
  __bf16* wprojT = (__bf16*)(ws + WS_WPROJT);
  __bf16* qb     = (__bf16*)(ws + WS_QBF);
  __bf16* kb     = (__bf16*)(ws + WS_KBF);
  __bf16* vtb    = (__bf16*)(ws + WS_VTB);
  __bf16* aout   = (__bf16*)(ws + WS_AOUT);

  const int nx = MTOT * EMB;
  convert_f32_bf16<<<(nx + 255) / 256, 256, 0, stream>>>(x, xbf, nx);
  transpose_f32_bf16<<<(EMB * NQKV + 255) / 256, 256, 0, stream>>>(Wqkv, wqkvT, EMB, NQKV);
  transpose_f32_bf16<<<(EMB * EMB + 255) / 256, 256, 0, stream>>>(Wproj, wprojT, EMB, EMB);

  qkv_gemm_kernel<<<dim3(NQKV / 256, MTOT / 128), 256, 0, stream>>>(
      xbf, wqkvT, bqkv, qb, kb, vtb);

  attn_kernel<<<BATCH * NH * (SEQ / 128), 128, 0, stream>>>(qb, kb, vtb, aout);

  proj_gemm_kernel<<<dim3(EMB / 256, MTOT / 128), 256, 0, stream>>>(
      aout, wprojT, bproj, out);
}
